// CgpHmmLayer_60206851556014
// MI455X (gfx1250) — matrix operand order repaired and checked by reading, not compiled, not run on hardware
//
#include <hip/hip_runtime.h>
#include <hip/hip_bf16.h>
#include <math.h>

// Problem constants (from reference)
#define S_STATES 305
#define SP       320     // padded states: multiple of 32 (K) and 16 (N)
#define KC       10      // K chunks of 32 -> 320
#define NT       20      // N tiles of 16 -> 320
#define BB       64
#define TT       4096

typedef __attribute__((ext_vector_type(16))) __bf16 v16bf;
typedef __attribute__((ext_vector_type(8)))  __bf16 v8bf;
typedef __attribute__((ext_vector_type(8)))  float  v8f;

// Workspace layout (bytes)
#define WS_OBS 0
#define WS_APK (WS_OBS + (size_t)BB * TT * sizeof(int))          // obs indices [B,T]
#define WS_BM  (WS_APK + (size_t)SP * SP * sizeof(__bf16))       // packed bf16 A
#define WS_I   (WS_BM  + (size_t)SP * 4  * sizeof(float))        // Bm softmax [SP,4]
#define WS_END (WS_I   + (size_t)SP * sizeof(float))             // I softmax [SP]

// ---------------- setup: one-hot -> obs index ----------------
__global__ void obs_kernel(const float* __restrict__ onehot, int* __restrict__ obs) {
  int i = blockIdx.x * blockDim.x + threadIdx.x;
  if (i < BB * TT) {
    const float* p = onehot + (size_t)i * 4;
    float v = p[1] + 2.0f * p[2] + 3.0f * p[3];   // dot(one_hot, [0,1,2,3])
    obs[i] = (int)(v + 0.5f);
  }
}

// ---------------- setup: transition softmax, written in WMMA-B fragment order ----
// B-fragment layout used here (contiguous K; 32x16 bf16, 8 VGPRs/lane): lane l -> column n = l%16,
// elements e=0..15 hold K = 16*(l/16) + e. Packed index:
//   apk[ ((tile*KC + kc)*32 + lane)*16 + e ]  with s = 32*kc + 16*half + e, n = 16*tile + l%16
__global__ void packA_kernel(const float* __restrict__ tk, __bf16* __restrict__ apk) {
  int s = threadIdx.x;               // 0..319, one padded row each
  int kc = s >> 5, K = s & 31, half = K >> 4, e = K & 15;
  if (s < S_STATES) {
    const float* row = tk + (size_t)s * S_STATES;
    float mx = row[0];
    for (int n = 1; n < S_STATES; ++n) mx = fmaxf(mx, row[n]);
    float sum = 0.0f;
    for (int n = 0; n < S_STATES; ++n) sum += __expf(row[n] - mx);
    float inv = 1.0f / sum;
    for (int n = 0; n < SP; ++n) {
      float v = (n < S_STATES) ? __expf(row[n] - mx) * inv : 0.0f;
      int tile = n >> 4;
      int lane = (half << 4) | (n & 15);
      apk[(((tile * KC + kc) * 32 + lane) << 4) + e] = (__bf16)v;
    }
  } else {  // pad rows -> 0
    for (int n = 0; n < SP; ++n) {
      int tile = n >> 4;
      int lane = (half << 4) | (n & 15);
      apk[(((tile * KC + kc) * 32 + lane) << 4) + e] = (__bf16)0.0f;
    }
  }
}

// ---------------- setup: emission + init softmax ----------------
__global__ void emitI_kernel(const float* __restrict__ ek, const float* __restrict__ ik,
                             float* __restrict__ bm, float* __restrict__ iv) {
  int s = threadIdx.x;  // 0..319
  if (s < S_STATES) {
    float a0 = ek[s*4+0], a1 = ek[s*4+1], a2 = ek[s*4+2], a3 = ek[s*4+3];
    float mx = fmaxf(fmaxf(a0, a1), fmaxf(a2, a3));
    float e0 = __expf(a0-mx), e1 = __expf(a1-mx), e2 = __expf(a2-mx), e3 = __expf(a3-mx);
    float inv = 1.0f / (e0 + e1 + e2 + e3);
    bm[s*4+0] = e0*inv; bm[s*4+1] = e1*inv; bm[s*4+2] = e2*inv; bm[s*4+3] = e3*inv;
  } else {
    bm[s*4+0] = 0.f; bm[s*4+1] = 0.f; bm[s*4+2] = 0.f; bm[s*4+3] = 0.f;
  }
  if (s == 0) {
    float mx = ik[0];
    for (int n = 1; n < S_STATES; ++n) mx = fmaxf(mx, ik[n]);
    float sum = 0.0f;
    for (int n = 0; n < S_STATES; ++n) sum += __expf(ik[n] - mx);
    float inv = 1.0f / sum;
    for (int n = 0; n < SP; ++n) iv[n] = (n < S_STATES) ? __expf(ik[n] - mx) * inv : 0.0f;
  }
}

// emission select: Bm row held in 4 registers, pick by 2-bit obs index (cndmask chain)
__device__ __forceinline__ float esel(int o, float b0, float b1, float b2, float b3) {
  float lo = (o & 1) ? b1 : b0;
  float hi = (o & 1) ? b3 : b2;
  return (o & 2) ? hi : lo;
}

// approximate reciprocal for alpha scaling (log-likelihood uses the sum itself)
__device__ __forceinline__ float fast_rcp(float x) {
#if __has_builtin(__builtin_amdgcn_rcpf)
  return __builtin_amdgcn_rcpf(x);
#else
  return 1.0f / x;
#endif
}

// load one K-chunk's WMMA fragments from LDS (2+2 x 16B loads)
__device__ __forceinline__ void load_frag(const __bf16* __restrict__ arow,
                                          const __bf16* __restrict__ sA,
                                          int tile, int kc, int half, int lane,
                                          v16bf& af, v16bf& bfg) {
  const int kb = kc * 32 + half * 16;  // A, contiguous K order: elems 0..15 = K[kb..kb+15] (matches packed B)
  v8bf alo = *(const v8bf*)(arow + kb);
  v8bf ahi = *(const v8bf*)(arow + kb + 8);
  const __bf16* bp = sA + (((tile * KC + kc) * 32 + lane) << 4);
  v8bf b0 = *(const v8bf*)bp;
  v8bf b1 = *(const v8bf*)(bp + 8);
  #pragma unroll
  for (int i = 0; i < 8; ++i) {
    af[i] = alo[i];  af[8 + i] = ahi[i];
    bfg[i] = b0[i];  bfg[8 + i] = b1[i];
  }
}

// scheduler hints: group 4 DS reads then 1 WMMA per pipeline stage
#if __has_builtin(__builtin_amdgcn_sched_group_barrier)
#define SCHED_DS4()  __builtin_amdgcn_sched_group_barrier(0x100, 4, 0)  // DS read x4
#define SCHED_WMMA() __builtin_amdgcn_sched_group_barrier(0x008, 1, 0)  // MFMA/WMMA x1
#else
#define SCHED_DS4()
#define SCHED_WMMA()
#endif

// CDNA5 split workgroup barrier (writer-side LDS drain + signal, separate wait)
__device__ __forceinline__ void bar_signal_drained() {
  asm volatile("s_wait_dscnt 0x0\n\ts_barrier_signal -1" ::: "memory");
}
__device__ __forceinline__ void bar_wait() {
  asm volatile("s_barrier_wait -1" ::: "memory");
}

// ---------------- main: 4 workgroups x 20 waves, full scan in LDS ----------------
__launch_bounds__(640, 1)
__global__ void hmm_forward_kernel(const int* __restrict__ obs,
                                   const __bf16* __restrict__ apk,
                                   const float* __restrict__ bm,
                                   const float* __restrict__ iv,
                                   float* __restrict__ out) {
  extern __shared__ char smem[];
  __bf16* sA     = (__bf16*)smem;                                   // SP*SP bf16   (204800 B)
  __bf16* sAlpha = (__bf16*)(smem + (size_t)SP * SP * 2);           // 16*SP bf16   (10240 B)
  float*  sRow   = (float*)(smem + (size_t)SP * SP * 2 + 16*SP*2);  // 2*16 f32
  int*    sObs   = (int*)((char*)sRow + 2 * 16 * sizeof(float));    // 2*16 int

  const int tid  = threadIdx.x;
  const int wave = tid >> 5;          // 0..19 -> N tile
  const int lane = tid & 31;
  const int half = lane >> 4;
  const int nloc = lane & 15;
  const int tile = wave;
  const int n    = tile * 16 + nloc;  // this lane's output state column
  const int mb   = blockIdx.x * 16;   // batch base for this workgroup

  // ---- one-time cooperative staging into LDS ----
  {
    const uint4* src = (const uint4*)apk;
    uint4* dst = (uint4*)sA;
    const int n16 = (SP * SP * 2) / 16;          // 12800 uint4
    for (int i = tid; i < n16; i += blockDim.x) dst[i] = src[i];
    if (tid < 16) {
      sObs[tid]      = obs[(size_t)(mb + tid) * TT + 0];
      sObs[16 + tid] = obs[(size_t)(mb + tid) * TT + 1];
    }
    if (tid < 32) sRow[tid] = 0.0f;
  }
  // per-lane constants: emission row Bm[n, 0..3] and init prob I[n]
  const float bm0 = bm[n * 4 + 0], bm1 = bm[n * 4 + 1];
  const float bm2 = bm[n * 4 + 2], bm3 = bm[n * 4 + 3];
  const float ivn = iv[n];
  __syncthreads();

  float ll = 0.0f;        // valid on wave 0, lanes 0..15 (one batch row each)
  const __bf16* arow = sAlpha + (size_t)nloc * SP;   // A-operand: row m = lane%16

  // shared per-step epilogue: row sums -> normalize -> store bf16 alpha -> log
  auto epilogue = [&](float (&acc)[8], int t) {
    const int buf = t & 1;
    // wave 0: prefetch next step's obs; clear the other row-sum buffer for t+1
    if (wave == 0 && lane < 16) {
      if (t + 1 < TT) sObs[((t + 1) & 1) * 16 + lane] = obs[(size_t)(mb + lane) * TT + (t + 1)];
      if (t > 0)      sRow[(1 - buf) * 16 + lane] = 0.0f;
    }
    // per-row partial sums: butterfly across the 16 lanes of each half, then LDS atomic
    #pragma unroll
    for (int r = 0; r < 8; ++r) {
      float p = acc[r];
      p += __shfl_xor(p, 1, 16);
      p += __shfl_xor(p, 2, 16);
      p += __shfl_xor(p, 4, 16);
      p += __shfl_xor(p, 8, 16);
      if (nloc == 0) atomicAdd(&sRow[buf * 16 + r + 8 * half], p);
    }
    __syncthreads();   // B1: row sums complete
    // normalize, write bf16 alpha for next step
    #pragma unroll
    for (int r = 0; r < 8; ++r) {
      int m = r + 8 * half;
      float inv = fast_rcp(sRow[buf * 16 + m]);
      sAlpha[(size_t)m * SP + n] = (__bf16)(acc[r] * inv);
    }
    // B2 as split barrier: drain LDS stores + signal, overlap wave0's log, then wait.
    // Safe: after B2 only wave0 itself touches sRow[buf] again (its own clear at t+1).
    bar_signal_drained();
    if (wave == 0 && lane < 16) ll += __logf(sRow[buf * 16 + lane]);
    bar_wait();
  };

  // ---- t = 0 (peeled): alpha0 = I * E0 ----
  {
    int4 o0 = *(const int4*)(sObs + 8 * half);
    int4 o1 = *(const int4*)(sObs + 8 * half + 4);
    int om[8] = {o0.x, o0.y, o0.z, o0.w, o1.x, o1.y, o1.z, o1.w};
    float acc[8];
    #pragma unroll
    for (int r = 0; r < 8; ++r)
      acc[r] = ivn * esel(om[r], bm0, bm1, bm2, bm3);
    epilogue(acc, 0);
  }

  // ---- steady state: t = 1 .. T-1 ----
  for (int t = 1; t < TT; ++t) {
    const int buf = t & 1;
    // this lane's 8 row-obs indices (m = 8*half .. 8*half+7): two 16B LDS loads
    int4 o0 = *(const int4*)(sObs + buf * 16 + 8 * half);
    int4 o1 = *(const int4*)(sObs + buf * 16 + 8 * half + 4);
    int om[8] = {o0.x, o0.y, o0.z, o0.w, o1.x, o1.y, o1.z, o1.w};

    // alpha_new = (alpha @ A): 10 x v_wmma_f32_16x16x32_bf16, distance-2 pipelined
    v8f c = {0.f, 0.f, 0.f, 0.f, 0.f, 0.f, 0.f, 0.f};
    v16bf afb[3], bfb[3];
    load_frag(arow, sA, tile, 0, half, lane, afb[0], bfb[0]);
    load_frag(arow, sA, tile, 1, half, lane, afb[1], bfb[1]);
    #pragma unroll
    for (int kc = 0; kc < KC; ++kc) {
      const int cur = kc % 3;
      if (kc + 2 < KC) {
        load_frag(arow, sA, tile, kc + 2, half, lane,
                  afb[(kc + 2) % 3], bfb[(kc + 2) % 3]);
        SCHED_DS4();
      }
      c = __builtin_amdgcn_wmma_f32_16x16x32_bf16(false, afb[cur], false, bfb[cur],
                                                  (short)0, c, false, false);
      SCHED_WMMA();
    }
    // emission multiply: a[m,n] *= Bm[n, obs[m,t]]
    float acc[8];
    #pragma unroll
    for (int r = 0; r < 8; ++r)
      acc[r] = c[r] * esel(om[r], bm0, bm1, bm2, bm3);

    epilogue(acc, t);
  }

  if (wave == 0 && lane < 16) out[mb + lane] = ll;
}

extern "C" void kernel_launch(void* const* d_in, const int* in_sizes, int n_in,
                              void* d_out, int out_size, void* d_ws, size_t ws_size,
                              hipStream_t stream) {
  (void)in_sizes; (void)n_in; (void)out_size; (void)ws_size;
  const float* onehot = (const float*)d_in[0];   // [B,T,4] one-hot f32
  const float* ik     = (const float*)d_in[1];   // [S]
  const float* tk     = (const float*)d_in[2];   // [S,S]
  const float* ek     = (const float*)d_in[3];   // [S,4]
  float* out = (float*)d_out;

  char* ws = (char*)d_ws;
  int*    obs = (int*)(ws + WS_OBS);
  __bf16* apk = (__bf16*)(ws + WS_APK);
  float*  bmv = (float*)(ws + WS_BM);
  float*  ivv = (float*)(ws + WS_I);

  obs_kernel<<<(BB * TT + 255) / 256, 256, 0, stream>>>(onehot, obs);
  packA_kernel<<<1, SP, 0, stream>>>(tk, apk);
  emitI_kernel<<<1, SP, 0, stream>>>(ek, ik, bmv, ivv);

  const size_t smemBytes = (size_t)SP * SP * 2   // packed A (bf16)
                         + (size_t)16 * SP * 2   // alpha (bf16)
                         + 2 * 16 * 4            // row sums (double buffered)
                         + 2 * 16 * 4;           // obs (double buffered)
  (void)hipFuncSetAttribute((const void*)hmm_forward_kernel,
                            hipFuncAttributeMaxDynamicSharedMemorySize, (int)smemBytes);
  hmm_forward_kernel<<<4, 640, smemBytes, stream>>>(obs, apk, bmv, ivv, out);
}
